// CWFA_AO_84232898609627
// MI455X (gfx1250) — matrix operand order repaired and checked by reading, not compiled, not run on hardware
//
#include <hip/hip_runtime.h>
#include <hip/hip_bf16.h>

// CDNA5 / gfx1250, wave32.
typedef _Float16 v16h __attribute__((ext_vector_type(16)));
typedef _Float16 v8h  __attribute__((ext_vector_type(8)));
typedef float    v8f  __attribute__((ext_vector_type(8)));

#define RANK   32
#define ENC    32
#define HID    64
#define BATCH  128
#define TSEQ   512
#define TC     64                 // t-chunk size (keeps C chunk at 16MB)
#define NCHUNK (TSEQ / TC)

// ---------------------------------------------------------------------------
// Kernel 1: two 2-layer MLP encoders (LeakyReLU -> sigmoid), output f16.
// ---------------------------------------------------------------------------
__device__ __forceinline__ void encode_one(const float* __restrict__ x,
    const float* __restrict__ W0, const float* __restrict__ b0,
    const float* __restrict__ W1, const float* __restrict__ b1,
    _Float16* __restrict__ outp)
{
  float xv[16];
#pragma unroll
  for (int j = 0; j < 16; j++) xv[j] = x[j];
  float h[64];
#pragma unroll 4
  for (int i = 0; i < 64; i++) {
    float s = b0[i];
#pragma unroll
    for (int j = 0; j < 16; j++) s += xv[j] * W0[j * 64 + i];
    h[i] = (s >= 0.0f) ? s : 0.01f * s;     // LeakyReLU(0.01)
  }
#pragma unroll 2
  for (int e = 0; e < 32; e++) {
    float s = b1[e];
#pragma unroll 8
    for (int k = 0; k < 64; k++) s += h[k] * W1[k * 32 + e];
    outp[e] = (_Float16)(1.0f / (1.0f + __expf(-s)));   // sigmoid
  }
}

__global__ __launch_bounds__(256) void cwfa_enc_kernel(
    const float* __restrict__ act, const float* __restrict__ obs,
    const float* __restrict__ aW0, const float* __restrict__ ab0,
    const float* __restrict__ aW1, const float* __restrict__ ab1,
    const float* __restrict__ oW0, const float* __restrict__ ob0,
    const float* __restrict__ oW1, const float* __restrict__ ob1,
    _Float16* __restrict__ aenc, _Float16* __restrict__ oenc)
{
  __shared__ float sw[6336];   // all 8 weight tensors, f32
  const int tid = threadIdx.x;
  for (int i = tid; i < 1024; i += 256) sw[i]        = aW0[i];
  for (int i = tid; i < 64;   i += 256) sw[1024 + i] = ab0[i];
  for (int i = tid; i < 2048; i += 256) sw[1088 + i] = aW1[i];
  for (int i = tid; i < 32;   i += 256) sw[3136 + i] = ab1[i];
  for (int i = tid; i < 1024; i += 256) sw[3168 + i] = oW0[i];
  for (int i = tid; i < 64;   i += 256) sw[4192 + i] = ob0[i];
  for (int i = tid; i < 2048; i += 256) sw[4256 + i] = oW1[i];
  for (int i = tid; i < 32;   i += 256) sw[6304 + i] = ob1[i];
  __syncthreads();

  const int token = blockIdx.x * 256 + tid;   // 0 .. B*T-1
  encode_one(act + (size_t)token * 16, sw,        sw + 1024, sw + 1088, sw + 3136,
             aenc + (size_t)token * 32);
  encode_one(obs + (size_t)token * 16, sw + 3168, sw + 4192, sw + 4256, sw + 6304,
             oenc + (size_t)token * 32);
}

// ---------------------------------------------------------------------------
// Kernel 2: repack A[p,j,k,l] (f32) -> A_mat[(j*32+k), (p*32+l)] (f16)
// ---------------------------------------------------------------------------
__global__ __launch_bounds__(256) void cwfa_convA_kernel(
    const float* __restrict__ A, _Float16* __restrict__ Am)
{
  const int idx = blockIdx.x * 256 + threadIdx.x;   // 0 .. 1M-1 (output index)
  const int pl = idx & 1023, jk = idx >> 10;
  const int j = jk >> 5, k = jk & 31, p = pl >> 5, l = pl & 31;
  Am[idx] = (_Float16)A[(((p * 32 + j) * 32 + k) * 32) + l];
}

// ---------------------------------------------------------------------------
// Kernel 3: WMMA GEMM  C[m, p*32+l] = sum_{jk} (a[m,j]*o[m,k]) * Amat[jk, pl]
// 128x128 block tile, 8 waves (4M x 2N), each wave 2 M-tiles x 4 N-tiles.
// A-fragments synthesized in registers (outer product a_j * o_k, v_pk_mul_f16).
// A_mat K-panels double-buffered in LDS (paired-k b32 transpose stores);
// one barrier per K-iteration. C written back via LDS repack -> b128 stores.
// ---------------------------------------------------------------------------
__global__ __launch_bounds__(256) void cwfa_gemm_kernel(
    const _Float16* __restrict__ aenc, const _Float16* __restrict__ oenc,
    const _Float16* __restrict__ Am, _Float16* __restrict__ Cc, int chunk)
{
  __shared__ __align__(16)  _Float16 sA[128][32];      // a[m][j]
  __shared__ __align__(16)  _Float16 sO[128][32];      // o[m][k]
  __shared__ __align__(128) _Float16 sB[2][128][64];   // Amat panel, [n][k], x2

  const int tid   = threadIdx.x;
  const int lane  = tid & 31;
  const int wave  = tid >> 5;
  const int waveM = wave & 3;          // 0..3  -> 32-row strip
  const int waveN = wave >> 2;         // 0..1  -> 64-col strip
  const int mblk  = blockIdx.x;        // 64 blocks of 128 rows (2 batches)
  const int n0    = blockIdx.y * 128;  // 8 col panels

  // per-thread panel staging indices: 2 iters x (2 adjacent k-rows, 8 cols)
  int qK[2], qC8[2];
#pragma unroll
  for (int it = 0; it < 2; it++) {
    int cidx = it * 256 + tid;         // 0..511
    qK[it]   = (cidx >> 4) * 2;        // 0,2,..,62
    qC8[it]  = (cidx & 15) * 8;        // 0..120
  }

  // ---- stage a / o tiles ----
#pragma unroll
  for (int t = 0; t < 2; t++) {
    int idx  = tid + t * 256;          // 0..511 (16B units, 4 per 64B row)
    int row  = idx >> 2;
    int part = idx & 3;
    int b    = mblk * 2 + (row >> 6);
    int tc   = row & 63;
    long tok = (long)b * TSEQ + (long)chunk * TC + tc;
    ((uint4*)&sA[row][0])[part] = *((const uint4*)(aenc + tok * 32) + part);
    ((uint4*)&sO[row][0])[part] = *((const uint4*)(oenc + tok * 32) + part);
  }

  // ---- prologue: stage K-panel 0 into sB[0] ----
  uint4 panLo[2], panHi[2];
#pragma unroll
  for (int it = 0; it < 2; it++) {
    panLo[it] = *(const uint4*)(Am + (size_t)qK[it] * 1024 + n0 + qC8[it]);
    panHi[it] = *(const uint4*)(Am + (size_t)(qK[it] + 1) * 1024 + n0 + qC8[it]);
  }
#pragma unroll
  for (int it = 0; it < 2; it++) {
    const unsigned short* lo = (const unsigned short*)&panLo[it];
    const unsigned short* hi = (const unsigned short*)&panHi[it];
#pragma unroll
    for (int i = 0; i < 8; i++) {
      unsigned v = (unsigned)lo[i] | ((unsigned)hi[i] << 16);   // (k, k+1) pair
      *(unsigned*)&sB[0][qC8[it] + i][qK[it]] = v;              // ds_store_b32
    }
  }
  __syncthreads();

  // Per-lane o sub-vectors (contiguous K layout, matching B):
  // lanes 0-15 hold K=0..15; lanes 16-31 hold K=16..31.
  const int kb0 = (lane >> 4) * 16;
  const int rl0 = waveM * 32 + (lane & 15);
  v8h olo[2], ohi[2];
#pragma unroll
  for (int mt = 0; mt < 2; mt++) {
    olo[mt] = *(const v8h*)&sO[rl0 + mt * 16][kb0];
    ohi[mt] = *(const v8h*)&sO[rl0 + mt * 16][kb0 + 8];
  }

  v8f acc[2][4];
#pragma unroll
  for (int mt = 0; mt < 2; mt++)
#pragma unroll
    for (int nt = 0; nt < 4; nt++)
#pragma unroll
      for (int i = 0; i < 8; i++) acc[mt][nt][i] = 0.0f;

  const int nrow0 = waveN * 64 + (lane & 15);
  const int kfsel = (lane >> 4) * 16;

  // ---- K loop: 16 panels of KC=64, double-buffered, 1 barrier/iter ----
  int cur = 0;
  for (int kk = 0; kk < 16; kk++) {
    // issue next panel's global loads (latency hidden under WMMA)
    if (kk < 15) {
#pragma unroll
      for (int it = 0; it < 2; it++) {
        size_t base = (size_t)((kk + 1) * 64 + qK[it]) * 1024 + n0 + qC8[it];
        panLo[it] = *(const uint4*)(Am + base);
        panHi[it] = *(const uint4*)(Am + base + 1024);
      }
      if (kk < 14) {   // prefetch panel kk+2 (global_prefetch_b8)
        const _Float16* pf = Am +
            (size_t)(kk * 64 + 128 + (tid & 63)) * 1024 + n0 + (tid >> 6) * 32;
        __builtin_prefetch(pf, 0, 0);
      }
    }

#pragma unroll
    for (int sl = 0; sl < 2; sl++) {
      const int j = kk * 2 + sl;        // ao K-chunk of 32 <=> fixed j
      // A-fragments: packed f16 outer-product rescale (v_pk_mul_f16)
      v16h af[2];
#pragma unroll
      for (int mt = 0; mt < 2; mt++) {
        _Float16 aval = sA[rl0 + mt * 16][j];
        v8h plo = olo[mt] * aval;
        v8h phi = ohi[mt] * aval;
        af[mt] = __builtin_shufflevector(plo, phi,
                  0, 1, 2, 3, 4, 5, 6, 7, 8, 9, 10, 11, 12, 13, 14, 15);
      }
      // B-fragments: preload all four, then burst 8 WMMAs
      v16h bf[4];
#pragma unroll
      for (int nt = 0; nt < 4; nt++)
        bf[nt] = *(const v16h*)&sB[cur][nrow0 + nt * 16][sl * 32 + kfsel];
#pragma unroll
      for (int nt = 0; nt < 4; nt++)
#pragma unroll
        for (int mt = 0; mt < 2; mt++)
          acc[mt][nt] = __builtin_amdgcn_wmma_f32_16x16x32_f16(
              false, af[mt], false, bf[nt], (short)0, acc[mt][nt], false, false);
    }

    // store next panel into the alternate buffer (paired-k b32 stores)
    if (kk < 15) {
#pragma unroll
      for (int it = 0; it < 2; it++) {
        const unsigned short* lo = (const unsigned short*)&panLo[it];
        const unsigned short* hi = (const unsigned short*)&panHi[it];
#pragma unroll
        for (int i = 0; i < 8; i++) {
          unsigned v = (unsigned)lo[i] | ((unsigned)hi[i] << 16);
          *(unsigned*)&sB[cur ^ 1][qC8[it] + i][qK[it]] = v;
        }
      }
    }
    __syncthreads();
    cur ^= 1;
  }

  // ---- writeback: repack through LDS, then coalesced b128 global stores ----
  _Float16* sC = (_Float16*)&sB[0][0][0];   // reuse 32KB (K-loop ended on barrier)
#pragma unroll
  for (int mt = 0; mt < 2; mt++) {
#pragma unroll
    for (int nt = 0; nt < 4; nt++) {
#pragma unroll
      for (int r = 0; r < 8; r++) {
        int rowLocal = waveM * 32 + mt * 16 + r + (lane >> 4) * 8;
        int colLocal = waveN * 64 + nt * 16 + (lane & 15);
        sC[rowLocal * 128 + colLocal] = (_Float16)acc[mt][nt][r];
      }
    }
  }
  __syncthreads();
  {
    int row  = tid >> 1;                 // 0..127
    int half = tid & 1;                  // 64-col half
    const uint4* src = (const uint4*)(sC + row * 128 + half * 64);
    uint4* dst = (uint4*)(Cc + ((size_t)mblk * 128 + row) * 1024 + n0 + half * 64);
#pragma unroll
    for (int i = 0; i < 8; i++) dst[i] = src[i];   // 128B per thread, coalesced
  }
}

// ---------------------------------------------------------------------------
// Kernel 4: sequential chain  h_t = h_{t-1} @ C_t   (one wave per batch),
// ping-pong h in LDS; final chunk also applies Omega and writes out[b].
// ---------------------------------------------------------------------------
__global__ __launch_bounds__(32) void cwfa_scan_kernel(
    const _Float16* __restrict__ Cc, const float* __restrict__ alpha,
    const float* __restrict__ Omega, float* __restrict__ hstate,
    float* __restrict__ outp, int chunk)
{
  __shared__ float h[2][32];
  __shared__ float red[32];
  const int b = blockIdx.x, lane = threadIdx.x;

  h[0][lane] = (chunk == 0) ? alpha[lane] : hstate[b * 32 + lane];
  __syncthreads();

  int cur = 0;
  for (int tc = 0; tc < TC; tc++) {
    const _Float16* row = Cc + ((size_t)(b * TC + tc) << 10);
    if (tc + 4 < TC)      // speculative prefetch 4 steps ahead
      __builtin_prefetch(row + 4 * 1024 + lane * 32, 0, 0);
    float acc = 0.0f;
#pragma unroll 8
    for (int p = 0; p < 32; p++)
      acc += h[cur][p] * (float)row[p * 32 + lane];   // coalesced across lanes
    h[cur ^ 1][lane] = acc;
    __syncthreads();
    cur ^= 1;
  }

  if (chunk == NCHUNK - 1) {
    red[lane] = h[cur][lane] * Omega[lane];
    __syncthreads();
    if (lane == 0) {
      float s = 0.0f;
      for (int p = 0; p < 32; p++) s += red[p];
      outp[b] = s;
    }
  } else {
    hstate[b * 32 + lane] = h[cur][lane];
  }
}

// ---------------------------------------------------------------------------
extern "C" void kernel_launch(void* const* d_in, const int* in_sizes, int n_in,
                              void* d_out, int out_size, void* d_ws, size_t ws_size,
                              hipStream_t stream)
{
  (void)in_sizes; (void)n_in; (void)out_size; (void)ws_size;
  const float* action = (const float*)d_in[0];
  const float* obs    = (const float*)d_in[1];
  const float* alpha  = (const float*)d_in[2];
  const float* A      = (const float*)d_in[3];
  const float* Omega  = (const float*)d_in[4];
  const float* aW0 = (const float*)d_in[5];
  const float* ab0 = (const float*)d_in[6];
  const float* aW1 = (const float*)d_in[7];
  const float* ab1 = (const float*)d_in[8];
  const float* oW0 = (const float*)d_in[9];
  const float* ob0 = (const float*)d_in[10];
  const float* oW1 = (const float*)d_in[11];
  const float* ob1 = (const float*)d_in[12];
  float* outp = (float*)d_out;

  char* ws = (char*)d_ws;
  const size_t NT = (size_t)BATCH * TSEQ;          // 65536 tokens
  _Float16* aenc   = (_Float16*)(ws);                               // 4 MB
  _Float16* oenc   = (_Float16*)(ws + NT * 32 * 2);                 // 4 MB
  _Float16* AmatH  = (_Float16*)(ws + NT * 32 * 2 * 2);             // 2 MB
  float*    hstate = (float*)   (ws + NT * 32 * 2 * 2 + (1024 * 1024 * 2));
  _Float16* Cc     = (_Float16*)(ws + NT * 32 * 2 * 2 + (1024 * 1024 * 2) + 65536); // 16 MB

  cwfa_enc_kernel<<<(unsigned)(NT / 256), 256, 0, stream>>>(
      action, obs, aW0, ab0, aW1, ab1, oW0, ob0, oW1, ob1, aenc, oenc);
  cwfa_convA_kernel<<<(1024 * 1024) / 256, 256, 0, stream>>>(A, AmatH);

  for (int c = 0; c < NCHUNK; c++) {
    cwfa_gemm_kernel<<<dim3(BATCH / 2, 8), 256, 0, stream>>>(aenc, oenc, AmatH, Cc, c);
    cwfa_scan_kernel<<<BATCH, 32, 0, stream>>>(Cc, alpha, Omega, hstate, outp, c);
  }
}
